// MultiHeadAttention_30760555774185
// MI455X (gfx1250) — matrix operand order repaired and checked by reading, not compiled, not run on hardware
//
#include <hip/hip_runtime.h>

// MHA forward for MI455X (gfx1250, wave32).  All GEMMs via v_wmma_f32_16x16x32_bf16.
// Round 2: all fp32->bf16 conversion hoisted out of GEMM inner loops into a
// one-shot elementwise kernel; GEMM loops are pure b128 loads + WMMA.
// Pipeline: cvt(q,k,v,w*) -> proj(q,k,v) -> flash attention (causal) -> out proj.

#define DMODEL 1024
#define NHEAD  16
#define DKH    64
#define BATCH  2
#define SEQ    2048
#define ROWS   (BATCH * SEQ)   // 4096

typedef __attribute__((ext_vector_type(16))) __bf16 v16bf;
typedef __attribute__((ext_vector_type(8)))  float  v8f;
typedef unsigned short u16;
typedef unsigned int   u32;

// native bf16 convert (RNE) -- lowers to HW cvt on gfx1250
__device__ __forceinline__ u16 f2bf(float f) {
  __bf16 h = (__bf16)f;
  return __builtin_bit_cast(u16, h);
}

union ABv {
  v16bf v;
  u16   s[16];
  uint4 q[2];
};

__device__ __forceinline__ v8f wmma_bf16(v16bf a, v16bf b, v8f c) {
  // (neg_a, A, neg_b, B, c_mod, C, reuse_a, reuse_b)
  return __builtin_amdgcn_wmma_f32_16x16x32_bf16(false, a, false, b, (short)0, c, false, false);
}

// ---------------------------------------------------------------------------
// One-shot fp32 -> bf16 conversion, 4 elements/thread, bandwidth-bound.
// ---------------------------------------------------------------------------
__global__ __launch_bounds__(256)
void cvt_kernel(const float* __restrict__ X, u16* __restrict__ Y, int n4) {
  const int i = blockIdx.x * 256 + threadIdx.x;
  if (i >= n4) return;
  const float4 f = reinterpret_cast<const float4*>(X)[i];
  ushort4 o;
  o.x = f2bf(f.x); o.y = f2bf(f.y); o.z = f2bf(f.z); o.w = f2bf(f.w);
  reinterpret_cast<ushort4*>(Y)[i] = o;
}

// ---------------------------------------------------------------------------
// Projection: Y = X @ W^T  (X, W preconverted bf16).  One wave: 32x64 tile.
// vt_mode==0: Y row-major bf16 [ROWS x DMODEL]
// vt_mode==1: Y transposed per head: Yt[((b*NHEAD+h)*DKH + d)*SEQ + s]
// ---------------------------------------------------------------------------
__global__ __launch_bounds__(32)
void proj_kernel(const u16* __restrict__ X, const u16* __restrict__ W,
                 u16* __restrict__ Y, int vt_mode) {
  const int lane = threadIdx.x & 31;
  const int g  = lane >> 4;     // lane half: selects K sub-range
  const int ml = lane & 15;     // M (A) / N (B) index

  const int Mb = blockIdx.x * 32;
  const int Nb = blockIdx.y * 64;

  v8f acc[2][4];
#pragma unroll
  for (int mh = 0; mh < 2; ++mh)
#pragma unroll
    for (int t = 0; t < 4; ++t)
#pragma unroll
      for (int i = 0; i < 8; ++i) acc[mh][t][i] = 0.0f;

  const u16* xrow0 = X + (size_t)(Mb + ml) * DMODEL;
  const u16* xrow1 = X + (size_t)(Mb + 16 + ml) * DMODEL;

  for (int kk = 0; kk < DMODEL; kk += 32) {
    // A tiles (16x32 bf16): lane M=ml; s[0..7]->K=8g.., s[8..15]->K=16+8g..
    ABv a0, a1;
    a0.q[0] = *reinterpret_cast<const uint4*>(xrow0 + kk + 8 * g);
    a0.q[1] = *reinterpret_cast<const uint4*>(xrow0 + kk + 16 + 8 * g);
    a1.q[0] = *reinterpret_cast<const uint4*>(xrow1 + kk + 8 * g);
    a1.q[1] = *reinterpret_cast<const uint4*>(xrow1 + kk + 16 + 8 * g);
#pragma unroll
    for (int t = 0; t < 4; ++t) {
      // B tile (32x16): B[k][n] = W[n][k]; lane N=ml, K = {8g..8g+7, 16+8g..16+8g+7}
      const u16* wrow = W + (size_t)(Nb + t * 16 + ml) * DMODEL + kk + 8 * g;
      ABv bm;
      bm.q[0] = *reinterpret_cast<const uint4*>(wrow);
      bm.q[1] = *reinterpret_cast<const uint4*>(wrow + 16);
      acc[0][t] = wmma_bf16(a0.v, bm.v, acc[0][t]);
      acc[1][t] = wmma_bf16(a1.v, bm.v, acc[1][t]);
    }
  }

  // C layout: vgpr r -> row M = r + 8g, col N = ml
#pragma unroll
  for (int mh = 0; mh < 2; ++mh) {
#pragma unroll
    for (int t = 0; t < 4; ++t) {
#pragma unroll
      for (int r = 0; r < 8; ++r) {
        const int row = Mb + 16 * mh + r + 8 * g;
        const int col = Nb + t * 16 + ml;
        const u16 hv = f2bf(acc[mh][t][r]);
        if (!vt_mode) {
          Y[(size_t)row * DMODEL + col] = hv;
        } else {
          const int b = row >> 11, s = row & (SEQ - 1);
          const int h = col >> 6,  d = col & (DKH - 1);
          Y[((size_t)(b * NHEAD + h) * DKH + d) * SEQ + s] = hv;
        }
      }
    }
  }
}

// ---------------------------------------------------------------------------
// Flash attention, causal.  One wave per (16 query rows, (b,h)).
// Q,K: bf16 row-major [ROWS x DMODEL]; Vt: bf16 [(b,h,dk,s)]; O: bf16 row-major.
// ---------------------------------------------------------------------------
__global__ __launch_bounds__(32)
void attn_kernel(const u16* __restrict__ Q, const u16* __restrict__ K,
                 const u16* __restrict__ Vt, u16* __restrict__ O) {
  __shared__ alignas(16) u16 ldsP[16 * 32];   // P tile, row-major 16 x 32

  const int lane = threadIdx.x & 31;
  const int g  = lane >> 4;
  const int ml = lane & 15;
  const int bh = blockIdx.y;
  const int b  = bh >> 4;
  const int h  = bh & 15;
  const int q0 = blockIdx.x * 16;

  // Q A-tiles: 16x64 as two 16x32 A matrices, kept in registers for all key blocks
  ABv aq[2];
  {
    const u16* qrow = Q + (size_t)(b * SEQ + q0 + ml) * DMODEL + h * DKH;
#pragma unroll
    for (int kk = 0; kk < 2; ++kk) {
      aq[kk].q[0] = *reinterpret_cast<const uint4*>(qrow + kk * 32 + 8 * g);
      aq[kk].q[1] = *reinterpret_cast<const uint4*>(qrow + kk * 32 + 16 + 8 * g);
    }
  }

  float mrun[8], lrun[8];
#pragma unroll
  for (int r = 0; r < 8; ++r) { mrun[r] = -1e30f; lrun[r] = 0.0f; }
  v8f o[4];
#pragma unroll
  for (int t = 0; t < 4; ++t)
#pragma unroll
    for (int i = 0; i < 8; ++i) o[t][i] = 0.0f;

  const int nkb = q0 / 32 + 1;   // causal: last needed 32-key block
  for (int j = 0; j < nkb; ++j) {
    const int k0 = j * 32;

    // scores S = Q K^T: two 16x16 C tiles (keys k0..k0+15, k0+16..k0+31)
    v8f sc[2];
#pragma unroll
    for (int t = 0; t < 2; ++t) {
#pragma unroll
      for (int i = 0; i < 8; ++i) sc[t][i] = 0.0f;
      const u16* krow = K + (size_t)(b * SEQ + k0 + t * 16 + ml) * DMODEL + h * DKH + 8 * g;
#pragma unroll
      for (int kk = 0; kk < 2; ++kk) {
        ABv bk;   // B[kfeat][n] = K[key n][feat]; lane N=ml, K = {8g..8g+7, 16+8g..16+8g+7}
        bk.q[0] = *reinterpret_cast<const uint4*>(krow + kk * 32);
        bk.q[1] = *reinterpret_cast<const uint4*>(krow + kk * 32 + 16);
        sc[t] = wmma_bf16(aq[kk].v, bk.v, sc[t]);
      }
    }

    // online softmax; C rows (M = r + 8g) live in one 16-lane half -> shfl_xor 1..8
#pragma unroll
    for (int r = 0; r < 8; ++r) {
      const int qi = q0 + r + 8 * g;
      float v0 = sc[0][r] * 0.125f;                 // 1/sqrt(64)
      float v1 = sc[1][r] * 0.125f;
      if (k0 + ml      > qi) v0 = -1e9f;            // reference NEG
      if (k0 + 16 + ml > qi) v1 = -1e9f;
      float rmax = fmaxf(v0, v1);
#pragma unroll
      for (int d = 1; d < 16; d <<= 1) rmax = fmaxf(rmax, __shfl_xor(rmax, d, 32));
      const float nm   = fmaxf(mrun[r], rmax);
      const float corr = __expf(mrun[r] - nm);
      const float p0   = __expf(v0 - nm);
      const float p1   = __expf(v1 - nm);
      float rsum = p0 + p1;
#pragma unroll
      for (int d = 1; d < 16; d <<= 1) rsum += __shfl_xor(rsum, d, 32);
      lrun[r] = lrun[r] * corr + rsum;
      mrun[r] = nm;
#pragma unroll
      for (int t = 0; t < 4; ++t) o[t][r] *= corr;
      ldsP[(r + 8 * g) * 32 + ml]      = f2bf(p0);
      ldsP[(r + 8 * g) * 32 + 16 + ml] = f2bf(p1);
    }
    __syncthreads();

    // reload P as A-matrix (16x32): lane M=ml; K = {8g..8g+7, 16+8g..16+8g+7}
    ABv pa;
    pa.q[0] = *reinterpret_cast<const uint4*>(&ldsP[ml * 32 + 8 * g]);
    pa.q[1] = *reinterpret_cast<const uint4*>(&ldsP[ml * 32 + 16 + 8 * g]);

    // O += P @ V: 4 feature tiles; Vt gives contiguous keys per feature row
#pragma unroll
    for (int t = 0; t < 4; ++t) {
      const u16* vrow = Vt + ((size_t)bh * DKH + t * 16 + ml) * SEQ + k0 + 8 * g;
      ABv bv;
      bv.q[0] = *reinterpret_cast<const uint4*>(vrow);
      bv.q[1] = *reinterpret_cast<const uint4*>(vrow + 16);
      o[t] = wmma_bf16(pa.v, bv.v, o[t]);
    }
    __syncthreads();
  }

  // normalize and store (merge heads = write into head's column slice)
#pragma unroll
  for (int t = 0; t < 4; ++t) {
#pragma unroll
    for (int r = 0; r < 8; ++r) {
      const int row = b * SEQ + q0 + r + 8 * g;
      const int col = h * DKH + t * 16 + ml;
      O[(size_t)row * DMODEL + col] = f2bf(o[t][r] / lrun[r]);
    }
  }
}

// ---------------------------------------------------------------------------
// Output projection: out = O @ w_o^T  (O bf16, w_o preconverted bf16, out fp32)
// One wave: 32x64 tile.
// ---------------------------------------------------------------------------
__global__ __launch_bounds__(32)
void oproj_kernel(const u16* __restrict__ X, const u16* __restrict__ W,
                  float* __restrict__ Y) {
  const int lane = threadIdx.x & 31;
  const int g  = lane >> 4;
  const int ml = lane & 15;
  const int Mb = blockIdx.x * 32;
  const int Nb = blockIdx.y * 64;

  v8f acc[2][4];
#pragma unroll
  for (int mh = 0; mh < 2; ++mh)
#pragma unroll
    for (int t = 0; t < 4; ++t)
#pragma unroll
      for (int i = 0; i < 8; ++i) acc[mh][t][i] = 0.0f;

  const u16* xrow0 = X + (size_t)(Mb + ml) * DMODEL;
  const u16* xrow1 = X + (size_t)(Mb + 16 + ml) * DMODEL;

  for (int kk = 0; kk < DMODEL; kk += 32) {
    ABv a0, a1;
    a0.q[0] = *reinterpret_cast<const uint4*>(xrow0 + kk + 8 * g);
    a0.q[1] = *reinterpret_cast<const uint4*>(xrow0 + kk + 16 + 8 * g);
    a1.q[0] = *reinterpret_cast<const uint4*>(xrow1 + kk + 8 * g);
    a1.q[1] = *reinterpret_cast<const uint4*>(xrow1 + kk + 16 + 8 * g);
#pragma unroll
    for (int t = 0; t < 4; ++t) {
      const u16* wrow = W + (size_t)(Nb + t * 16 + ml) * DMODEL + kk + 8 * g;
      ABv bm;
      bm.q[0] = *reinterpret_cast<const uint4*>(wrow);
      bm.q[1] = *reinterpret_cast<const uint4*>(wrow + 16);
      acc[0][t] = wmma_bf16(a0.v, bm.v, acc[0][t]);
      acc[1][t] = wmma_bf16(a1.v, bm.v, acc[1][t]);
    }
  }

#pragma unroll
  for (int mh = 0; mh < 2; ++mh)
#pragma unroll
    for (int t = 0; t < 4; ++t)
#pragma unroll
      for (int r = 0; r < 8; ++r)
        Y[(size_t)(Mb + 16 * mh + r + 8 * g) * DMODEL + Nb + t * 16 + ml] = acc[mh][t][r];
}

// ---------------------------------------------------------------------------
extern "C" void kernel_launch(void* const* d_in, const int* in_sizes, int n_in,
                              void* d_out, int out_size, void* d_ws, size_t ws_size,
                              hipStream_t stream) {
  (void)in_sizes; (void)n_in; (void)out_size; (void)ws_size;
  const float* q  = (const float*)d_in[0];
  const float* k  = (const float*)d_in[1];
  const float* v  = (const float*)d_in[2];
  // d_in[3] = causal mask; applied analytically in attn_kernel
  const float* wq = (const float*)d_in[4];
  const float* wk = (const float*)d_in[5];
  const float* wv = (const float*)d_in[6];
  const float* wo = (const float*)d_in[7];

  const size_t N  = (size_t)ROWS * DMODEL;    // 4M activation elements
  const size_t Nw = (size_t)DMODEL * DMODEL;  // 1M weight elements

  u16* qb  = (u16*)d_ws;        // bf16 copies of inputs
  u16* kb  = qb + N;
  u16* vb  = kb + N;
  u16* Qb  = vb + N;            // projected Q/K, V transposed, attention out
  u16* Kb  = Qb + N;
  u16* Vt  = Kb + N;
  u16* Ob  = Vt + N;
  u16* wqb = Ob + N;            // bf16 weights
  u16* wkb = wqb + Nw;
  u16* wvb = wkb + Nw;
  u16* wob = wvb + Nw;

  const int na4 = (int)(N / 4), nw4 = (int)(Nw / 4);
  cvt_kernel<<<dim3((na4 + 255) / 256), dim3(256), 0, stream>>>(q,  qb,  na4);
  cvt_kernel<<<dim3((na4 + 255) / 256), dim3(256), 0, stream>>>(k,  kb,  na4);
  cvt_kernel<<<dim3((na4 + 255) / 256), dim3(256), 0, stream>>>(v,  vb,  na4);
  cvt_kernel<<<dim3((nw4 + 255) / 256), dim3(256), 0, stream>>>(wq, wqb, nw4);
  cvt_kernel<<<dim3((nw4 + 255) / 256), dim3(256), 0, stream>>>(wk, wkb, nw4);
  cvt_kernel<<<dim3((nw4 + 255) / 256), dim3(256), 0, stream>>>(wv, wvb, nw4);
  cvt_kernel<<<dim3((nw4 + 255) / 256), dim3(256), 0, stream>>>(wo, wob, nw4);

  dim3 gp(ROWS / 32, DMODEL / 64);
  dim3 bl(32);
  proj_kernel<<<gp, bl, 0, stream>>>(qb, wqb, Qb, 0);
  proj_kernel<<<gp, bl, 0, stream>>>(kb, wkb, Kb, 0);
  proj_kernel<<<gp, bl, 0, stream>>>(vb, wvb, Vt, 1);
  attn_kernel<<<dim3(SEQ / 16, BATCH * NHEAD), bl, 0, stream>>>(Qb, Kb, Vt, Ob);
  oproj_kernel<<<gp, bl, 0, stream>>>(Ob, wob, (float*)d_out);
}
